// XCABlock_31009663877554
// MI455X (gfx1250) — compile-verified, matrix operand order checked by reading, not run on hardware
//
#include <hip/hip_runtime.h>
#include <hip/hip_bf16.h>
#include <math.h>

// ---------------------------------------------------------------------------
// XCA block for MI455X (gfx1250, wave32, WMMA).
// Shapes: B=8, H=W=64, N=4096, C=384, nh=8, d=48, hid=1536.
// Matmuls on v_wmma_f32_16x16x32_f16; GEMM tiles staged to LDS with
// global_load_async_to_lds_b128 (ASYNCcnt) and double-buffered.
// ---------------------------------------------------------------------------

typedef __attribute__((ext_vector_type(16))) _Float16 v16h;
typedef __attribute__((ext_vector_type(8)))  _Float16 v8h;
typedef __attribute__((ext_vector_type(8)))  float    v8f;

#define BB   8
#define HH   64
#define WW   64
#define NN   4096          // H*W
#define CC   384
#define NHD  8             // heads
#define DD   48            // C / nh
#define HIDN 1536
#define ROWS (BB * NN)     // 32768

__device__ __forceinline__ float gelu_exact(float x) {
    return 0.5f * x * (1.0f + erff(x * 0.70710678118654752f));
}

__device__ __forceinline__ v8f wmma16x16x32(v16h a, v16h b, v8f c) {
    return __builtin_amdgcn_wmma_f32_16x16x32_f16(
        /*neg_a=*/false, a, /*neg_b=*/false, b,
        /*c_mod=*/(short)0, c, /*reuse_a=*/false, /*reuse_b=*/false);
}

// Low 32 bits of a flat pointer into the LDS aperture == LDS byte offset.
__device__ __forceinline__ unsigned lds_off_of(const void* p) {
    return (unsigned)(uintptr_t)p;
}

// Async copy of 16 bytes (per lane): GLOBAL[saddr + voff] -> LDS[lds].
// GVS addressing; tracked by ASYNCcnt.
__device__ __forceinline__ void async_cp16(unsigned lds, unsigned goff, const void* base) {
    asm volatile("global_load_async_to_lds_b128 %0, %1, %2"
                 :: "v"(lds), "v"(goff), "s"(base) : "memory");
}
// Same with +16B on BOTH the LDS and global address (ISA: INST_OFFSET added to both).
__device__ __forceinline__ void async_cp16_o16(unsigned lds, unsigned goff, const void* base) {
    asm volatile("global_load_async_to_lds_b128 %0, %1, %2 offset:16"
                 :: "v"(lds), "v"(goff), "s"(base) : "memory");
}
__device__ __forceinline__ void wait_async_le4() {
    asm volatile("s_wait_asynccnt 0x4" ::: "memory");
}
__device__ __forceinline__ void wait_async_le0() {
    asm volatile("s_wait_asynccnt 0x0" ::: "memory");
}

// Load one 16x32 f16 WMMA A/B fragment from a K-contiguous (NT) matrix.
// Per-lane layout (wave32): lane<16 row=lane holds K = {0..7, 16..23},
// lane>=16 row=lane-16 holds K = {8..15, 24..31}. Two 16-byte loads.
__device__ __forceinline__ v16h load_frag(const _Float16* p, int ld, int r0,
                                          int k0, int lane) {
    const int half = lane >> 4;
    const int row  = lane & 15;
    const _Float16* base = p + (size_t)(r0 + row) * ld + k0 + half * 8;
    v8h lo = *(const v8h*)(base);
    v8h hi = *(const v8h*)(base + 16);
    v16h f;
#pragma unroll
    for (int j = 0; j < 8; ++j) { f[j] = lo[j]; f[j + 8] = hi[j]; }
    return f;
}

// ---------------------------------------------------------------------------
// fp32 -> f16 weight conversion
// ---------------------------------------------------------------------------
__global__ void k_cvt_f16(const float* __restrict__ in, _Float16* __restrict__ out, int n) {
    int i = blockIdx.x * blockDim.x + threadIdx.x;
    if (i < n) out[i] = (_Float16)in[i];
}

// ---------------------------------------------------------------------------
// LayerNorm over C=384 per row; f16 out (GEMM feed) or f32 out (LPI branch).
// ---------------------------------------------------------------------------
template <typename OT>
__global__ void k_layernorm(const float* __restrict__ x, const float* __restrict__ w,
                            const float* __restrict__ b, OT* __restrict__ out) {
    const int row = blockIdx.x;
    const int t   = threadIdx.x;
    const float* xr = x + (size_t)row * CC;
    float s = 0.f, s2 = 0.f;
    for (int c = t; c < CC; c += 128) { float v = xr[c]; s += v; s2 += v * v; }
    __shared__ float r1[128], r2[128];
    r1[t] = s; r2[t] = s2; __syncthreads();
    for (int off = 64; off > 0; off >>= 1) {
        if (t < off) { r1[t] += r1[t + off]; r2[t] += r2[t + off]; }
        __syncthreads();
    }
    const float mu = r1[0] * (1.0f / CC);
    const float var = r2[0] * (1.0f / CC) - mu * mu;
    const float rs = rsqrtf(var + 1e-5f);
    OT* orow = out + (size_t)row * CC;
    for (int c = t; c < CC; c += 128)
        orow[c] = (OT)((xr[c] - mu) * rs * w[c] + b[c]);
}

// ---------------------------------------------------------------------------
// Tiled NT WMMA GEMM: out[m,f] = sum_k A[m,k] * W[f,k]  (+ epilogue)
// 128x128 tile per 256-thread block (8 waves, 2x4 16x16 tiles per wave).
// K staged 32-wide via async-to-LDS, double-buffered / software-pipelined.
// EPI: 0 = (+bias) -> f16 ; 1 = gelu(+bias) -> f16 ; 2 = resid + gamma*(+bias) -> f32
// ---------------------------------------------------------------------------
template <int EPI>
__global__ __launch_bounds__(256)
void k_gemm_nt(const _Float16* __restrict__ A, const _Float16* __restrict__ W,
               const float* __restrict__ bias, const float* __restrict__ gamma,
               const float* __restrict__ resid, void* __restrict__ outp,
               int Nf, int K) {
    __shared__ _Float16 sA[2][128 * 40];   // pad stride 40 halves (80B)
    __shared__ _Float16 sW[2][128 * 40];
    const int tid  = threadIdx.x;
    const int lane = tid & 31;
    const int wid  = tid >> 5;
    const int wm   = wid & 3;          // 4 wave-rows of 32
    const int wn   = wid >> 2;         // 2 wave-cols of 64
    const int n0   = blockIdx.x * 128;
    const int m0   = blockIdx.y * 128;

    v8f acc[2][4];
#pragma unroll
    for (int i = 0; i < 2; ++i)
#pragma unroll
        for (int j = 0; j < 4; ++j)
#pragma unroll
            for (int e = 0; e < 8; ++e) acc[i][j][e] = 0.f;

    const int lr  = tid >> 1;          // 0..127 tile row
    const int seg = (tid & 1) * 16;    // half-row of 16 halves

    // Per-lane byte offsets into A / W for this thread's 32-byte slice.
    unsigned aoff = (unsigned)(((size_t)(m0 + lr) * K + seg) * 2);
    unsigned woff = (unsigned)(((size_t)(n0 + lr) * K + seg) * 2);
    unsigned ldsA[2], ldsW[2];
    ldsA[0] = lds_off_of(&sA[0][lr * 40 + seg]);
    ldsA[1] = lds_off_of(&sA[1][lr * 40 + seg]);
    ldsW[0] = lds_off_of(&sW[0][lr * 40 + seg]);
    ldsW[1] = lds_off_of(&sW[1][lr * 40 + seg]);

    // Prologue: issue chunk 0 into buffer 0 (4 async instructions / wave).
    async_cp16(ldsA[0], aoff, A); async_cp16_o16(ldsA[0], aoff, A);
    async_cp16(ldsW[0], woff, W); async_cp16_o16(ldsW[0], woff, W);

    int ib = 0;
    for (int kc = 0; kc < K; kc += 32) {
        if (kc + 32 < K) {
            aoff += 64; woff += 64;    // next 32-halves chunk
            async_cp16(ldsA[ib ^ 1], aoff, A); async_cp16_o16(ldsA[ib ^ 1], aoff, A);
            async_cp16(ldsW[ib ^ 1], woff, W); async_cp16_o16(ldsW[ib ^ 1], woff, W);
            wait_async_le4();          // in-order: current chunk has landed
        } else {
            wait_async_le0();
        }
        __syncthreads();               // all waves' current chunk visible

        v16h af[2];
#pragma unroll
        for (int i = 0; i < 2; ++i)
            af[i] = load_frag(&sA[ib][0], 40, wm * 32 + i * 16, 0, lane);
        v16h bf[4];
#pragma unroll
        for (int j = 0; j < 4; ++j)
            bf[j] = load_frag(&sW[ib][0], 40, wn * 64 + j * 16, 0, lane);
#pragma unroll
        for (int j = 0; j < 4; ++j)
#pragma unroll
            for (int i = 0; i < 2; ++i)
                acc[i][j] = wmma16x16x32(af[i], bf[j], acc[i][j]);

        __syncthreads();               // done reading buf ib -> safe to refill
        ib ^= 1;
    }

    // Epilogue. C/D layout: lane<16 -> n=lane, m=r ; lane>=16 -> n=lane-16, m=8+r
#pragma unroll
    for (int i = 0; i < 2; ++i)
#pragma unroll
        for (int j = 0; j < 4; ++j)
#pragma unroll
            for (int rr = 0; rr < 8; ++rr) {
                const int gm = m0 + wm * 32 + i * 16 + (lane >> 4) * 8 + rr;
                const int gn = n0 + wn * 64 + j * 16 + (lane & 15);
                float v = acc[i][j][rr] + (bias ? bias[gn] : 0.f);
                const size_t o = (size_t)gm * Nf + gn;
                if (EPI == 0)      ((_Float16*)outp)[o] = (_Float16)v;
                else if (EPI == 1) ((_Float16*)outp)[o] = (_Float16)gelu_exact(v);
                else               ((float*)outp)[o] = resid[o] + gamma[gn] * v;
            }
}

// ---------------------------------------------------------------------------
// Column L2 norms of q and k: norms[b, f] = max(||qkv[b, :, f]||, 1e-12).
// ---------------------------------------------------------------------------
__global__ void k_colnorm(const _Float16* __restrict__ qkv, float* __restrict__ norms) {
    const int b = blockIdx.x / (2 * CC);
    const int f = blockIdx.x % (2 * CC);
    const _Float16* p = qkv + (size_t)b * NN * (3 * CC) + f;
    float s = 0.f;
    for (int n = threadIdx.x; n < NN; n += 256) {
        float v = (float)p[(size_t)n * (3 * CC)];
        s += v * v;
    }
    __shared__ float red[256];
    red[threadIdx.x] = s; __syncthreads();
    for (int off = 128; off > 0; off >>= 1) {
        if (threadIdx.x < off) red[threadIdx.x] += red[threadIdx.x + off];
        __syncthreads();
    }
    if (threadIdx.x == 0) norms[blockIdx.x] = fmaxf(sqrtf(red[0]), 1e-12f);
}

// ---------------------------------------------------------------------------
// Gram matrix G[b,h,i,j] = sum_n q[b,n,h*48+i] * k[b,n,C+h*48+j] via WMMA.
// One (b,h) per 96-thread block (3 waves = 3 m-tiles), K = N = 4096 staged
// through LDS in 32-wide chunks (transpose scatter -> synchronous staging).
// ---------------------------------------------------------------------------
__global__ __launch_bounds__(96)
void k_gram(const _Float16* __restrict__ qkv, float* __restrict__ G) {
    const int bh = blockIdx.x;
    const int b = bh >> 3, h = bh & 7;
    __shared__ _Float16 sq[DD * 40];
    __shared__ _Float16 sk[DD * 40];
    const int tid = threadIdx.x, lane = tid & 31, wv = tid >> 5;
    const int i0  = tid % DD;          // loop-invariant channel
    const int nn0 = tid / DD;          // 0 or 1
    const int qo  = h * DD + i0;
    const int ko  = CC + h * DD + i0;

    v8f acc[3];
#pragma unroll
    for (int j = 0; j < 3; ++j)
#pragma unroll
        for (int e = 0; e < 8; ++e) acc[j][e] = 0.f;

    for (int nc = 0; nc < NN; nc += 32) {
        __syncthreads();
#pragma unroll
        for (int it = 0; it < 16; ++it) {
            const int nn = nn0 + it * 2;
            const size_t rowb = (size_t)(b * NN + nc + nn) * (3 * CC);
            sq[i0 * 40 + nn] = qkv[rowb + qo];
            sk[i0 * 40 + nn] = qkv[rowb + ko];
        }
        __syncthreads();
        v16h af = load_frag(sq, 40, wv * 16, 0, lane);
#pragma unroll
        for (int j = 0; j < 3; ++j) {
            v16h bf = load_frag(sk, 40, j * 16, 0, lane);
            acc[j] = wmma16x16x32(af, bf, acc[j]);
        }
    }
#pragma unroll
    for (int j = 0; j < 3; ++j)
#pragma unroll
        for (int rr = 0; rr < 8; ++rr) {
            const int gm = wv * 16 + (lane >> 4) * 8 + rr;
            const int gn = j * 16 + (lane & 15);
            G[((size_t)bh * DD + gm) * DD + gn] = acc[j][rr];
        }
}

// ---------------------------------------------------------------------------
// Softmax over j (48) of temperature[h] * G / (||q_i|| * ||k_j||).
// Output f16, padded to 64 columns with zeros (so attn.V has K=64).
// ---------------------------------------------------------------------------
__global__ void k_softmax(const float* __restrict__ G, const float* __restrict__ norms,
                          const float* __restrict__ temp, _Float16* __restrict__ attn) {
    const int row = blockIdx.x;              // B*nh*48 rows
    const int bh = row / DD, i = row % DD;
    const int b = bh >> 3, h = bh & 7;
    const int j = threadIdx.x;               // 64 threads
    float v = -1e30f;
    if (j < DD) {
        const float nq = norms[b * (2 * CC) + h * DD + i];
        const float nk = norms[b * (2 * CC) + CC + h * DD + j];
        v = G[((size_t)bh * DD + i) * DD + j] * temp[h] / (nq * nk);
    }
    __shared__ float red[64];
    red[j] = v; __syncthreads();
    for (int off = 32; off > 0; off >>= 1) {
        if (j < off) red[j] = fmaxf(red[j], red[j + off]);
        __syncthreads();
    }
    const float mx = red[0]; __syncthreads();
    const float e = (j < DD) ? expf(v - mx) : 0.f;
    red[j] = e; __syncthreads();
    for (int off = 32; off > 0; off >>= 1) {
        if (j < off) red[j] += red[j + off];
        __syncthreads();
    }
    attn[(size_t)bh * (DD * 64) + i * 64 + j] = (_Float16)(e / red[0]);
}

// ---------------------------------------------------------------------------
// a[b,n,h*48+i] = sum_j attn[b,h,i,j] * v[b,n,2C+h*48+j]  via WMMA (K=64).
// Block: one (b,h) x 128-seq chunk; 4 waves cover 3x8 output tiles.
// Tiles staged with async-to-LDS (contiguous 16B slices per lane).
// ---------------------------------------------------------------------------
__global__ __launch_bounds__(128)
void k_attn_v(const _Float16* __restrict__ attn, const _Float16* __restrict__ qkv,
              _Float16* __restrict__ ah) {
    const int nt = blockIdx.x;               // 0..31 seq chunks of 128
    const int bh = blockIdx.y;
    const int b = bh >> 3, h = bh & 7;
    __shared__ _Float16 sa[DD * 64];         // attn tile (zero-padded K)
    __shared__ _Float16 sv[128 * 72];        // v^T tile, padded stride (144B rows)
    const int tid = threadIdx.x, lane = tid & 31, wv = tid >> 5;

    // attn tile: 6144 bytes flat; thread t copies 48 contiguous bytes.
    {
        unsigned lds = lds_off_of(sa) + tid * 48;
        unsigned go  = (unsigned)bh * (DD * 64 * 2) + tid * 48;
#pragma unroll
        for (int u = 0; u < 3; ++u) { async_cp16(lds, go, attn); lds += 16; go += 16; }
    }
    // v^T tile: thread t owns row t (48 halves = 96 contiguous bytes both sides).
    {
        unsigned lds = lds_off_of(sv) + tid * 144;
        unsigned go  = (unsigned)(((size_t)(b * NN + nt * 128 + tid) * (3 * CC)
                                   + 2 * CC + h * DD) * 2);
#pragma unroll
        for (int u = 0; u < 6; ++u) { async_cp16(lds, go, qkv); lds += 16; go += 16; }
    }
    // zero-pad columns 48..71 of own row (sync LDS stores).
    {
        v8h z = {};
        *(v8h*)&sv[tid * 72 + 48] = z;
        *(v8h*)&sv[tid * 72 + 56] = z;
        *(v8h*)&sv[tid * 72 + 64] = z;
    }
    wait_async_le0();
    __syncthreads();

#pragma unroll
    for (int s = 0; s < 6; ++s) {
        const int tt = wv * 6 + s;           // 24 tiles over 4 waves
        const int mt = tt / 8, ntile = tt % 8;
        v8f acc;
#pragma unroll
        for (int e = 0; e < 8; ++e) acc[e] = 0.f;
#pragma unroll
        for (int k0 = 0; k0 < 64; k0 += 32) {
            v16h af = load_frag(sa, 64, mt * 16, k0, lane);
            v16h bf = load_frag(sv, 72, ntile * 16, k0, lane);
            acc = wmma16x16x32(af, bf, acc);
        }
#pragma unroll
        for (int rr = 0; rr < 8; ++rr) {
            const int gm = mt * 16 + (lane >> 4) * 8 + rr;   // channel i
            const int gn = ntile * 16 + (lane & 15);         // seq within chunk
            ah[(size_t)(b * NN + nt * 128 + gn) * CC + h * DD + gm] = (_Float16)acc[rr];
        }
    }
}

// ---------------------------------------------------------------------------
// LPI branch: depthwise 3x3 conv (channel-last) + GELU; groupnorm stats;
// groupnorm-apply + depthwise conv + residual.
// ---------------------------------------------------------------------------
__global__ void k_dwconv_gelu(const float* __restrict__ in, const float* __restrict__ w,
                              const float* __restrict__ bias, float* __restrict__ out) {
    const int idx = blockIdx.x * blockDim.x + threadIdx.x;  // B*N*C exact
    const int c = idx % CC;
    const int sp = idx / CC;
    const int x = sp % WW, y = (sp / WW) % HH, b = sp / NN;
    float acc = 0.f;
#pragma unroll
    for (int ky = 0; ky < 3; ++ky) {
        const int yy = y + ky - 1;
        if (yy < 0 || yy >= HH) continue;
#pragma unroll
        for (int kx = 0; kx < 3; ++kx) {
            const int xx = x + kx - 1;
            if (xx < 0 || xx >= WW) continue;
            acc += in[(size_t)(b * NN + yy * WW + xx) * CC + c] * w[c * 9 + ky * 3 + kx];
        }
    }
    out[idx] = gelu_exact(acc + bias[c]);
}

__global__ void k_gn_stats(const float* __restrict__ y, float* __restrict__ stats) {
    const int bg = blockIdx.x;               // b*8+g
    const int b = bg >> 3, g = bg & 7;
    float s = 0.f, s2 = 0.f;
    for (int e = threadIdx.x; e < DD * NN; e += 256) {
        const int ch = g * DD + e % DD;
        const int sp = e / DD;
        const float v = y[(size_t)(b * NN + sp) * CC + ch];
        s += v; s2 += v * v;
    }
    __shared__ float r1[256], r2[256];
    r1[threadIdx.x] = s; r2[threadIdx.x] = s2; __syncthreads();
    for (int off = 128; off > 0; off >>= 1) {
        if (threadIdx.x < off) { r1[threadIdx.x] += r1[threadIdx.x + off];
                                 r2[threadIdx.x] += r2[threadIdx.x + off]; }
        __syncthreads();
    }
    if (threadIdx.x == 0) {
        const float inv = 1.0f / (DD * NN);
        const float mu = r1[0] * inv;
        const float var = r2[0] * inv - mu * mu;
        stats[bg * 2] = mu;
        stats[bg * 2 + 1] = rsqrtf(var + 1e-5f);
    }
}

__global__ void k_gn_dwconv_res(const float* __restrict__ y, const float* __restrict__ stats,
                                const float* __restrict__ gnw, const float* __restrict__ gnb,
                                const float* __restrict__ w, const float* __restrict__ bias,
                                const float* __restrict__ resid, const float* __restrict__ gamma,
                                float* __restrict__ out) {
    const int idx = blockIdx.x * blockDim.x + threadIdx.x;
    const int c = idx % CC;
    const int sp = idx / CC;
    const int x = sp % WW, yy0 = (sp / WW) % HH, b = sp / NN;
    const int g = c / DD;
    const float mu = stats[(b * 8 + g) * 2];
    const float inv = stats[(b * 8 + g) * 2 + 1];
    const float sc = inv * gnw[c];
    const float sh = gnb[c] - mu * sc;
    float acc = 0.f;
#pragma unroll
    for (int ky = 0; ky < 3; ++ky) {
        const int yy = yy0 + ky - 1;
        if (yy < 0 || yy >= HH) continue;
#pragma unroll
        for (int kx = 0; kx < 3; ++kx) {
            const int xx = x + kx - 1;
            if (xx < 0 || xx >= WW) continue;
            const float v = y[(size_t)(b * NN + yy * WW + xx) * CC + c] * sc + sh;
            acc += v * w[c * 9 + ky * 3 + kx];
        }
    }
    out[idx] = resid[idx] + gamma[c] * (acc + bias[c]);
}

// ---------------------------------------------------------------------------
// Host-side orchestration.
// ---------------------------------------------------------------------------
extern "C" void kernel_launch(void* const* d_in, const int* in_sizes, int n_in,
                              void* d_out, int out_size, void* d_ws, size_t ws_size,
                              hipStream_t stream) {
    const float* x      = (const float*)d_in[0];
    const float* n1_w   = (const float*)d_in[3];
    const float* n1_b   = (const float*)d_in[4];
    const float* qkv_w  = (const float*)d_in[5];
    const float* temp   = (const float*)d_in[6];
    const float* proj_w = (const float*)d_in[7];
    const float* proj_b = (const float*)d_in[8];
    const float* n3_w   = (const float*)d_in[9];
    const float* n3_b   = (const float*)d_in[10];
    const float* conv1w = (const float*)d_in[11];
    const float* conv1b = (const float*)d_in[12];
    const float* gn_w   = (const float*)d_in[13];
    const float* gn_b   = (const float*)d_in[14];
    const float* conv2w = (const float*)d_in[15];
    const float* conv2b = (const float*)d_in[16];
    const float* n2_w   = (const float*)d_in[17];
    const float* n2_b   = (const float*)d_in[18];
    const float* fc1_w  = (const float*)d_in[19];
    const float* fc1_b  = (const float*)d_in[20];
    const float* fc2_w  = (const float*)d_in[21];
    const float* fc2_b  = (const float*)d_in[22];
    const float* gamma1 = (const float*)d_in[23];
    const float* gamma2 = (const float*)d_in[24];
    const float* gamma3 = (const float*)d_in[25];

    char* ws = (char*)d_ws;
    size_t off = 0;
    auto carve = [&](size_t bytes) { char* p = ws + off; off += (bytes + 255) & ~(size_t)255; return p; };

    _Float16* wqkv_h = (_Float16*)carve((size_t)3 * CC * CC * 2);
    _Float16* wproj_h = (_Float16*)carve((size_t)CC * CC * 2);
    _Float16* wfc1_h = (_Float16*)carve((size_t)HIDN * CC * 2);
    _Float16* wfc2_h = (_Float16*)carve((size_t)CC * HIDN * 2);
    float*    norms  = (float*)carve((size_t)BB * 2 * CC * 4);
    float*    G      = (float*)carve((size_t)BB * NHD * DD * DD * 4);
    _Float16* attn   = (_Float16*)carve((size_t)BB * NHD * DD * 64 * 2);
    float*    stats  = (float*)carve((size_t)BB * 8 * 2 * 4);
    _Float16* lnh    = (_Float16*)carve((size_t)ROWS * CC * 2);          // f16 LN out
    _Float16* ah     = (_Float16*)carve((size_t)ROWS * CC * 2);          // attn output
    char*     big    = carve((size_t)ROWS * HIDN * 2);                   // phased region
    float*    x1     = (float*)carve((size_t)ROWS * CC * 4);
    // Phased aliases inside `big` (lifetimes are strictly sequential):
    _Float16* qkvh = (_Float16*)big;                                     // steps 3..7
    float*    lnf  = (float*)big;                                        // steps 9..10a
    float*    y1   = (float*)(big + (size_t)ROWS * CC * 4);              // steps 10a..10c
    _Float16* hbuf = (_Float16*)big;                                     // steps 12..13
    float*    outF = (float*)d_out;                                      // x2 lives here

    // 1) weights -> f16
    k_cvt_f16<<<(3 * CC * CC + 255) / 256, 256, 0, stream>>>(qkv_w, wqkv_h, 3 * CC * CC);
    k_cvt_f16<<<(CC * CC + 255) / 256, 256, 0, stream>>>(proj_w, wproj_h, CC * CC);
    k_cvt_f16<<<(HIDN * CC + 255) / 256, 256, 0, stream>>>(fc1_w, wfc1_h, HIDN * CC);
    k_cvt_f16<<<(CC * HIDN + 255) / 256, 256, 0, stream>>>(fc2_w, wfc2_h, CC * HIDN);

    // 2) LN1 -> f16
    k_layernorm<_Float16><<<ROWS, 128, 0, stream>>>(x, n1_w, n1_b, lnh);
    // 3) qkv GEMM (no bias) -> f16
    k_gemm_nt<0><<<dim3(3 * CC / 128, ROWS / 128), 256, 0, stream>>>(
        lnh, wqkv_h, nullptr, nullptr, nullptr, qkvh, 3 * CC, CC);
    // 4) q/k column norms
    k_colnorm<<<BB * 2 * CC, 256, 0, stream>>>(qkvh, norms);
    // 5) Gram matrices via WMMA over K=N
    k_gram<<<BB * NHD, 96, 0, stream>>>(qkvh, G);
    // 6) scaled softmax -> padded f16 attn
    k_softmax<<<BB * NHD * DD, 64, 0, stream>>>(G, norms, temp, attn);
    // 7) attn . V via WMMA
    k_attn_v<<<dim3(NN / 128, BB * NHD), 128, 0, stream>>>(attn, qkvh, ah);
    // 8) proj GEMM + residual: x1 = x + gamma1 * (a.projW^T + proj_b)
    k_gemm_nt<2><<<dim3(CC / 128, ROWS / 128), 256, 0, stream>>>(
        ah, wproj_h, proj_b, gamma1, x, x1, CC, CC);

    // 9) LN3 -> f32 (spatial branch)
    k_layernorm<float><<<ROWS, 128, 0, stream>>>(x1, n3_w, n3_b, lnf);
    // 10) LPI: dwconv+gelu, groupnorm stats, gn-apply+dwconv+residual -> x2 (= d_out)
    k_dwconv_gelu<<<ROWS * CC / 256, 256, 0, stream>>>(lnf, conv1w, conv1b, y1);
    k_gn_stats<<<BB * 8, 256, 0, stream>>>(y1, stats);
    k_gn_dwconv_res<<<ROWS * CC / 256, 256, 0, stream>>>(
        y1, stats, gn_w, gn_b, conv2w, conv2b, x1, gamma3, outF);

    // 11) LN2 -> f16
    k_layernorm<_Float16><<<ROWS, 128, 0, stream>>>(outF, n2_w, n2_b, lnh);
    // 12) fc1 GEMM + GELU -> f16
    k_gemm_nt<1><<<dim3(HIDN / 128, ROWS / 128), 256, 0, stream>>>(
        lnh, wfc1_h, fc1_b, nullptr, nullptr, hbuf, HIDN, CC);
    // 13) fc2 GEMM + residual in-place into d_out: out = x2 + gamma2*(h.fc2^T + b)
    k_gemm_nt<2><<<dim3(CC / 128, ROWS / 128), 256, 0, stream>>>(
        hbuf, wfc2_h, fc2_b, gamma2, outF, outF, CC, HIDN);

    (void)in_sizes; (void)n_in; (void)out_size; (void)ws_size;
}
